// GatedConv_12128987643930
// MI455X (gfx1250) — matrix operand order repaired and checked by reading, not compiled, not run on hardware
//
#include <hip/hip_runtime.h>
#include <hip/hip_bf16.h>

typedef __bf16 bf16_t;
typedef __attribute__((ext_vector_type(16))) bf16_t v16bf;
typedef __attribute__((ext_vector_type(8)))  bf16_t v8bf;
typedef __attribute__((ext_vector_type(8)))  float  v8f;

#define D 128
#define NUM_LAYERS 2

// ---------------------------------------------------------------- utilities
__global__ void zero_f32_kernel(float* __restrict__ p, int n) {
    int t = blockIdx.x * blockDim.x + threadIdx.x;
    if (t < n) p[t] = 0.0f;
}

__global__ void to_bf16_kernel(const float* __restrict__ in, bf16_t* __restrict__ out, int n) {
    int t = blockIdx.x * blockDim.x + threadIdx.x;
    if (t < n) out[t] = (bf16_t)in[t];
}

// Convert weights once: conv_w transposed per layer (so K is contiguous per
// output column), w_ih / w_hh are already [out, k] with k contiguous.
__global__ void convert_weights_kernel(const float* __restrict__ conv_w,
                                       const float* __restrict__ w_ih,
                                       const float* __restrict__ w_hh,
                                       bf16_t* __restrict__ convT,
                                       bf16_t* __restrict__ wihb,
                                       bf16_t* __restrict__ whhb) {
    int t = blockIdx.x * blockDim.x + threadIdx.x;
    const int CW = NUM_LAYERS * D * D;          // 32768
    const int GW = 3 * D * D;                   // 49152
    if (t < CW) {
        int l = t >> 14;            // t / (128*128)
        int n = (t >> 7) & 127;
        int k = t & 127;
        convT[t] = (bf16_t)conv_w[l * D * D + k * D + n];
    } else if (t < CW + GW) {
        int i = t - CW;
        wihb[i] = (bf16_t)w_ih[i];
    } else if (t < CW + 2 * GW) {
        int i = t - CW - GW;
        whhb[i] = (bf16_t)w_hh[i];
    }
}

// h[n,:] = embed[node_ids[n],:]
__global__ void gather_embed_kernel(const int* __restrict__ node_ids,
                                    const float* __restrict__ embed,
                                    float* __restrict__ h, int N) {
    int t = blockIdx.x * blockDim.x + threadIdx.x;
    if (t >= N * D) return;
    int n = t >> 7;
    int d = t & 127;
    h[t] = embed[(size_t)node_ids[n] * D + d];
}

// ------------------------------------------------------------- WMMA GEMM
// C[N,OUT] = A[N,128] x BT[OUT,128]^T   (i.e. C[n,o] = sum_k A[n,k]*BT[o,k])
// One wave = one 16x16 tile of C. 256 threads = 8 waves = 8 column tiles.
__global__ void __launch_bounds__(256)
wmma_gemm_bf16(const bf16_t* __restrict__ A,
               const bf16_t* __restrict__ BT,
               float* __restrict__ C,
               int N, int OUT) {
    const int lane = threadIdx.x & 31;
    const int wave = threadIdx.x >> 5;
    const int colTile = blockIdx.y * 8 + wave;
    const int col0 = colTile * 16;
    if (col0 >= OUT) return;                 // wave-uniform exit

    const int row0 = blockIdx.x << 4;
    const int half = lane >> 4;              // 0: lanes 0-15, 1: lanes 16-31
    const int l16  = lane & 15;
    int arow = row0 + l16;
    if (arow >= N) arow = N - 1;             // clamp for safe loads in a tail tile
    const int col = col0 + l16;

    const bf16_t* __restrict__ arowp = A  + (size_t)arow * D;
    const bf16_t* __restrict__ bcolp = BT + (size_t)col  * D;

    v8f acc = {};
#pragma unroll
    for (int kt = 0; kt < 4; ++kt) {
        const int kb = kt * 32;
        // A layout: lane = row, lanes 0-15 hold K kb..kb+15, lanes 16-31 K kb+16..kb+31
        v16bf a = *(const v16bf*)(arowp + kb + half * 16);
        // B layout: lane = column, lanes 0-15 hold K kb..kb+15, lanes 16-31 K kb+16..kb+31
        v16bf b = *(const v16bf*)(bcolp + kb + half * 16);
        acc = __builtin_amdgcn_wmma_f32_16x16x32_bf16(
            /*neg_a=*/false, a, /*neg_b=*/false, b,
            /*c_mod=*/(short)0, acc, /*reuse_a=*/false, /*reuse_b=*/false);
    }

    // D layout: VGPR r -> row (row0 + r) for lanes 0-15, (row0 + 8 + r) for lanes 16-31.
    // Tile fullness is wave-uniform: take the unconditional fast path for full
    // tiles (always the case when N % 16 == 0) so the 8 stores form one clause.
    float* __restrict__ cp = C + (size_t)(row0 + half * 8) * OUT + col;
    if (row0 + 16 <= N) {
#pragma unroll
        for (int r = 0; r < 8; ++r) {
            *cp = acc[r];
            cp += OUT;
        }
    } else {
        const int rowBase = row0 + half * 8;
#pragma unroll
        for (int r = 0; r < 8; ++r) {
            if (rowBase + r < N) *cp = acc[r];
            cp += OUT;
        }
    }
}

// -------------------------------------------------------------- scatter-add
// One wave per edge: lanes split the 128-float row into 4-float chunks.
// Edge indices are wave-uniform -> scalar loads; row read is one coalesced 512B burst.
__global__ void __launch_bounds__(256)
scatter_add_kernel(const float* __restrict__ m,
                   const int* __restrict__ src,
                   const int* __restrict__ dst,
                   float* __restrict__ agg, int E) {
    int t = blockIdx.x * blockDim.x + threadIdx.x;
    int e = t >> 5;
    int q = t & 31;                          // 32 chunks of 4 floats = 128
    if (e >= E) return;
    int s = src[e];
    int dnode = dst[e];
    const float4 v = *(const float4*)(m + (size_t)s * D + q * 4);
    float* out = agg + (size_t)dnode * D + q * 4;
    atomicAdd(out + 0, v.x);
    atomicAdd(out + 1, v.y);
    atomicAdd(out + 2, v.z);
    atomicAdd(out + 3, v.w);
}

// ------------------------------------------------------------------- GRU
__global__ void gru_kernel(const float* __restrict__ gi,
                           const float* __restrict__ gh,
                           float* __restrict__ h,
                           const float* __restrict__ b_ih,
                           const float* __restrict__ b_hh, int N) {
    int t = blockIdx.x * blockDim.x + threadIdx.x;
    if (t >= N * D) return;
    int n = t >> 7;
    int d = t & 127;
    size_t b3 = (size_t)n * (3 * D);
    float ir = gi[b3 + d]         + b_ih[d];
    float iz = gi[b3 + D + d]     + b_ih[D + d];
    float in = gi[b3 + 2 * D + d] + b_ih[2 * D + d];
    float hr = gh[b3 + d]         + b_hh[d];
    float hz = gh[b3 + D + d]     + b_hh[D + d];
    float hn = gh[b3 + 2 * D + d] + b_hh[2 * D + d];
    float r  = 1.0f / (1.0f + __expf(-(ir + hr)));
    float z  = 1.0f / (1.0f + __expf(-(iz + hz)));
    float ng = tanhf(in + r * hn);
    size_t hi = (size_t)n * D + d;
    h[hi] = (1.0f - z) * ng + z * h[hi];
}

// ------------------------------------------------------------------ pool
__global__ void pool_accum_kernel(const float* __restrict__ h,
                                  const int* __restrict__ batch,
                                  float* __restrict__ sums,
                                  float* __restrict__ counts, int N) {
    int t = blockIdx.x * blockDim.x + threadIdx.x;
    if (t >= N * D) return;
    int n = t >> 7;
    int d = t & 127;
    int g = batch[n];
    atomicAdd(&sums[(size_t)g * D + d], h[t]);
    if (d == 0) atomicAdd(&counts[g], 1.0f);
}

__global__ void pool_final_kernel(const float* __restrict__ sums,
                                  const float* __restrict__ counts,
                                  float* __restrict__ out, int G) {
    int t = blockIdx.x * blockDim.x + threadIdx.x;
    if (t >= G * D) return;
    int g = t >> 7;
    out[t] = sums[t] / fmaxf(counts[g], 1.0f);
}

// ------------------------------------------------------------------ launch
static inline int cdiv(int a, int b) { return (a + b - 1) / b; }

extern "C" void kernel_launch(void* const* d_in, const int* in_sizes, int n_in,
                              void* d_out, int out_size, void* d_ws, size_t ws_size,
                              hipStream_t stream) {
    const int*   node_ids   = (const int*)d_in[0];
    const int*   edge_index = (const int*)d_in[1];
    const int*   batch      = (const int*)d_in[2];
    // d_in[3] = num_graphs (device scalar); G derived from out_size instead.
    const float* embed      = (const float*)d_in[4];
    const float* conv_w     = (const float*)d_in[5];
    const float* w_ih       = (const float*)d_in[6];
    const float* w_hh       = (const float*)d_in[7];
    const float* b_ih       = (const float*)d_in[8];
    const float* b_hh       = (const float*)d_in[9];

    const int N = in_sizes[0];
    const int E = in_sizes[1] / 2;
    const int G = out_size / D;

    const int* src = edge_index;
    const int* dst = edge_index + E;

    // ---- carve workspace (256B aligned slots)
    char* ws = (char*)d_ws;
    size_t off = 0;
    auto alloc = [&](size_t bytes) -> char* {
        char* p = ws + off;
        off += (bytes + 255) & ~(size_t)255;
        return p;
    };
    float*  h      = (float*) alloc((size_t)N * D * 4);
    float*  m      = (float*) alloc((size_t)N * D * 4);
    float*  agg    = (float*) alloc((size_t)N * D * 4);
    float*  gi     = (float*) alloc((size_t)N * 3 * D * 4);
    float*  gh     = (float*) alloc((size_t)N * 3 * D * 4);
    bf16_t* h_bf   = (bf16_t*)alloc((size_t)N * D * 2);
    bf16_t* agg_bf = (bf16_t*)alloc((size_t)N * D * 2);
    bf16_t* convT  = (bf16_t*)alloc((size_t)NUM_LAYERS * D * D * 2);
    bf16_t* wihb   = (bf16_t*)alloc((size_t)3 * D * D * 2);
    bf16_t* whhb   = (bf16_t*)alloc((size_t)3 * D * D * 2);
    float*  sums   = (float*) alloc((size_t)G * D * 4 + (size_t)G * 4); // sums + counts contiguous
    float*  counts = sums + (size_t)G * D;

    const int BT = 256;
    const int nND  = N * D;
    const int wTot = NUM_LAYERS * D * D + 2 * (3 * D * D);

    convert_weights_kernel<<<cdiv(wTot, BT), BT, 0, stream>>>(conv_w, w_ih, w_hh, convT, wihb, whhb);
    gather_embed_kernel<<<cdiv(nND, BT), BT, 0, stream>>>(node_ids, embed, h, N);

    const int rowTiles = cdiv(N, 16);
    for (int l = 0; l < NUM_LAYERS; ++l) {
        // m = h @ conv_w[l]
        to_bf16_kernel<<<cdiv(nND, BT), BT, 0, stream>>>(h, h_bf, nND);
        wmma_gemm_bf16<<<dim3(rowTiles, 1), BT, 0, stream>>>(h_bf, convT + (size_t)l * D * D, m, N, D);
        // agg = segment_sum(m[src], dst)
        zero_f32_kernel<<<cdiv(nND, BT), BT, 0, stream>>>(agg, nND);
        scatter_add_kernel<<<cdiv(E * 32, BT), BT, 0, stream>>>(m, src, dst, agg, E);
        // GRU gates: gi = agg @ w_ih^T, gh = h @ w_hh^T  (biases added in gru_kernel)
        to_bf16_kernel<<<cdiv(nND, BT), BT, 0, stream>>>(agg, agg_bf, nND);
        wmma_gemm_bf16<<<dim3(rowTiles, 3), BT, 0, stream>>>(agg_bf, wihb, gi, N, 3 * D);
        wmma_gemm_bf16<<<dim3(rowTiles, 3), BT, 0, stream>>>(h_bf,   whhb, gh, N, 3 * D);
        gru_kernel<<<cdiv(nND, BT), BT, 0, stream>>>(gi, gh, h, b_ih, b_hh, N);
    }

    // global mean pool
    zero_f32_kernel<<<cdiv(G * D + G, BT), BT, 0, stream>>>(sums, G * D + G);
    pool_accum_kernel<<<cdiv(nND, BT), BT, 0, stream>>>(h, batch, sums, counts, N);
    pool_final_kernel<<<cdiv(G * D, BT), BT, 0, stream>>>(sums, counts, (float*)d_out, G);
}
